// SCP_Seq2Seq_3238405341654
// MI455X (gfx1250) — matrix operand order repaired and checked by reading, not compiled, not run on hardware
//
#include <hip/hip_runtime.h>
#include <stddef.h>

// ---------------- types ----------------
typedef __attribute__((ext_vector_type(8)))  float  v8f;
typedef __attribute__((ext_vector_type(16))) __bf16 v16bf;
typedef __attribute__((ext_vector_type(8)))  __bf16 v8bf;

union V16 { v16bf v; v8bf h[2]; };

#define Bsz 256
#define Tlen 512
#define Din 64
#define Hd 256
#define G4 1024
#define WOUT 96

// =====================================================================
// Core WMMA gate-GEMM + fused LSTM cell.
// Workgroup tile: M=64 (batch) x Jtile=32 (hidden units) => 4 gates * 32 = 128 gate cols.
// 4 waves (128 threads), wave w computes gate w (columns w*256 + j0 .. +31),
// K = K0 (input segment) + K1 (recurrent h segment), bf16 WMMA, f32 accum.
// After GEMM: exchange i/f/g/o via LDS, apply LSTM cell, write bf16 h (+copy) and f32 c.
// =====================================================================
__device__ __forceinline__ void lstm_gates_tile(
    const __bf16* __restrict__ act0, int act0_stride, int K0,
    const __bf16* __restrict__ act1, int K1,
    const __bf16* __restrict__ W, int Ktot,          // N=1024 rows x Ktot cols, row-major
    const float*  __restrict__ bias,                 // 1024
    float*  __restrict__ c_state,                    // B x 256 (in-place)
    __bf16* __restrict__ h_out,                      // B x 256
    __bf16* __restrict__ h_out2, int h_out2_stride)  // optional copy (enc_out slot)
{
  __shared__ float lds[4 * 64 * 32];   // [gate][m][j]  32KB
  const int tid   = threadIdx.x;
  const int wave  = tid >> 5;          // gate index 0..3 (i,f,g,o)
  const int lane  = tid & 31;
  const int lhalf = lane >> 4;         // 0/1
  const int lmod  = lane & 15;
  const int j0 = blockIdx.x * 32;      // hidden-unit tile base
  const int m0 = blockIdx.y * 64;      // batch tile base

  v8f acc[4][2];
#pragma unroll
  for (int mi = 0; mi < 4; ++mi)
#pragma unroll
    for (int ni = 0; ni < 2; ++ni)
      acc[mi][ni] = (v8f){0.f,0.f,0.f,0.f,0.f,0.f,0.f,0.f};

  for (int kc = 0; kc < Ktot; kc += 32) {
    const __bf16* src; int sstride, scol;
    if (kc < K0) { src = act0; sstride = act0_stride; scol = kc; }
    else         { src = act1; sstride = K1;          scol = kc - K0; }

    // A tiles: act[m0+mi*16 .. +15, kc..kc+31]
    // ISA 16-bit A 16x32 layout: lane L, elem e -> M=L%16, K=(e>=8?16:0)+(L/16)*8+(e&7)
    V16 A[4];
    const int ka = scol + lhalf * 8;
#pragma unroll
    for (int mi = 0; mi < 4; ++mi) {
      const int m = m0 + mi * 16 + lmod;
      const __bf16* p = src + (size_t)m * sstride + ka;
      A[mi].h[0] = *(const v8bf*)(p);
      A[mi].h[1] = *(const v8bf*)(p + 16);
    }
    // B tiles: W rows are B columns. lane L -> N=L%16, K=(e>=8?16:0)+(L/16)*8+(e&7)
    V16 Bm[2];
    const int kb = kc + lhalf * 8;
#pragma unroll
    for (int ni = 0; ni < 2; ++ni) {
      const int n = wave * 256 + j0 + ni * 16 + lmod;
      Bm[ni].h[0] = *(const v8bf*)(W + (size_t)n * Ktot + kb);
      Bm[ni].h[1] = *(const v8bf*)(W + (size_t)n * Ktot + kb + 16);
    }
#pragma unroll
    for (int mi = 0; mi < 4; ++mi)
#pragma unroll
      for (int ni = 0; ni < 2; ++ni)
        acc[mi][ni] = __builtin_amdgcn_wmma_f32_16x16x32_bf16(
            false, A[mi].v, false, Bm[ni].v, (short)0, acc[mi][ni], false, false);
  }

  // bias + stash to LDS.  C/D layout: lane L, vgpr v -> m=(L/16)*8+v, n=L%16
#pragma unroll
  for (int ni = 0; ni < 2; ++ni) {
    const float bv = bias[wave * 256 + j0 + ni * 16 + lmod];
    const int jl = ni * 16 + lmod;
#pragma unroll
    for (int mi = 0; mi < 4; ++mi)
#pragma unroll
      for (int v = 0; v < 8; ++v) {
        const int m = mi * 16 + lhalf * 8 + v;
        lds[(wave * 64 + m) * 32 + jl] = acc[mi][ni][v] + bv;
      }
  }
  __syncthreads();

  // fused LSTM cell
  for (int e = tid; e < 64 * 32; e += 128) {
    const int m = e >> 5, j = e & 31;
    const int b = m0 + m, jj = j0 + j;
    const float gi = lds[0 * 2048 + e];
    const float gf = lds[1 * 2048 + e];
    const float gg = lds[2 * 2048 + e];
    const float go = lds[3 * 2048 + e];
    const size_t ci = (size_t)b * 256 + jj;
    float c = c_state[ci];
    const float si = 1.f / (1.f + __expf(-gi));
    const float sf = 1.f / (1.f + __expf(-gf));
    const float so = 1.f / (1.f + __expf(-go));
    c = sf * c + si * tanhf(gg);
    const float h = so * tanhf(c);
    c_state[ci] = c;
    h_out[ci] = (__bf16)h;
    if (h_out2) h_out2[(size_t)b * h_out2_stride + jj] = (__bf16)h;
  }
}

// ---- encoder step: z=0 -> layer0 at t=s ; z=1 -> layer1 at t=s-1 (pipelined) ----
__global__ __launch_bounds__(128) void encoder_step_kernel(
    int s,
    const __bf16* __restrict__ x_bf,
    const __bf16* __restrict__ h0_read, __bf16* __restrict__ h0_write,
    const __bf16* __restrict__ h1_read, __bf16* __restrict__ h1_write,
    float* __restrict__ c0, float* __restrict__ c1,
    __bf16* __restrict__ enc_out,
    const __bf16* __restrict__ W0, const float* __restrict__ b0,
    const __bf16* __restrict__ W1, const float* __restrict__ b1)
{
  if (blockIdx.z == 0) {
    if (s >= Tlen) return;
    lstm_gates_tile(x_bf + (size_t)s * Din, Tlen * Din, Din,
                    h0_read, Hd, W0, Din + Hd, b0, c0, h0_write, nullptr, 0);
  } else {
    if (s == 0) return;
    lstm_gates_tile(h0_read, Hd, Hd,
                    h1_read, Hd, W1, 2 * Hd, b1, c1, h1_write,
                    enc_out + (size_t)(s - 1) * Hd, Tlen * Hd);
  }
}

__global__ __launch_bounds__(128) void decoder_gates_kernel(
    const __bf16* __restrict__ inp_bf,
    const __bf16* __restrict__ h_read, __bf16* __restrict__ h_write,
    float* __restrict__ c,
    const __bf16* __restrict__ Wd, const float* __restrict__ bd)
{
  lstm_gates_tile(inp_bf, 2 * Hd, 2 * Hd, h_read, Hd, Wd, 3 * Hd, bd, c,
                  h_write, nullptr, 0);
}

// ---- attention + input projection: one workgroup per batch element ----
__global__ __launch_bounds__(256) void attn_kernel(
    const __bf16* __restrict__ enc_out,   // B x T x H (bf16)
    const __bf16* __restrict__ h,         // B x H
    const float*  __restrict__ prev_y,    // B x 64
    const __bf16* __restrict__ Wp,        // 256 x 64
    const float*  __restrict__ bp,        // 256
    __bf16* __restrict__ inp_bf)          // B x 512 : [pin(256) | ctx(256)]
{
  __shared__ float sh[256];
  __shared__ float sc[512];
  __shared__ float red[256];
  __shared__ float sy[64];
  const int b = blockIdx.x, tid = threadIdx.x;
  sh[tid] = (float)h[(size_t)b * 256 + tid];
  if (tid < 64) sy[tid] = prev_y[b * 64 + tid];
  __syncthreads();

  // scores (2 timesteps per thread), scaled by 1/sqrt(H)=1/16
#pragma unroll
  for (int tt = 0; tt < 2; ++tt) {
    const int t = tid + tt * 256;
    const __bf16* row = enc_out + ((size_t)b * Tlen + t) * 256;
    float s = 0.f;
    for (int k = 0; k < 256; k += 8) {
      v8bf v = *(const v8bf*)(row + k);
#pragma unroll
      for (int u = 0; u < 8; ++u) s += (float)v[u] * sh[k + u];
    }
    sc[t] = s * 0.0625f;
  }
  __syncthreads();
  // softmax over T=512
  red[tid] = fmaxf(sc[tid], sc[tid + 256]);
  __syncthreads();
  for (int w = 128; w > 0; w >>= 1) {
    if (tid < w) red[tid] = fmaxf(red[tid], red[tid + w]);
    __syncthreads();
  }
  const float mx = red[0];
  __syncthreads();
  const float e0 = __expf(sc[tid] - mx), e1 = __expf(sc[tid + 256] - mx);
  sc[tid] = e0; sc[tid + 256] = e1;
  red[tid] = e0 + e1;
  __syncthreads();
  for (int w = 128; w > 0; w >>= 1) {
    if (tid < w) red[tid] += red[tid + w];
    __syncthreads();
  }
  const float inv = 1.f / red[0];
  __syncthreads();
  sc[tid] *= inv; sc[tid + 256] *= inv;
  __syncthreads();
  // ctx[k=tid] = sum_t w[t]*enc_out[b,t,k]  (coalesced over k)
  float acc = 0.f;
  for (int t = 0; t < Tlen; ++t)
    acc += sc[t] * (float)enc_out[((size_t)b * Tlen + t) * 256 + tid];
  inp_bf[(size_t)b * 512 + 256 + tid] = (__bf16)acc;
  // pin[j=tid] = bp[j] + prev_y . Wp[j,:]
  float pj = bp[tid];
  const __bf16* wr = Wp + tid * 64;
#pragma unroll 8
  for (int d = 0; d < 64; ++d) pj += sy[d] * (float)wr[d];
  inp_bf[(size_t)b * 512 + tid] = (__bf16)pj;
}

// ---- output projection y = [h;ctx] @ Wo^T + bo ; feeds prev_y ----
__global__ __launch_bounds__(64) void out_y_kernel(
    const __bf16* __restrict__ h,       // post-update h (B x 256)
    const __bf16* __restrict__ inp_bf,  // ctx lives at [b*512 + 256 ..)
    const __bf16* __restrict__ Wo,      // 64 x 512
    const float*  __restrict__ bo,      // 64
    float* __restrict__ out, int step,
    float* __restrict__ prev_y)
{
  __shared__ float v[512];
  const int b = blockIdx.x, tid = threadIdx.x;
  for (int k = tid; k < 512; k += 64)
    v[k] = (k < 256) ? (float)h[(size_t)b * 256 + k]
                     : (float)inp_bf[(size_t)b * 512 + k];
  __syncthreads();
  float acc = bo[tid];
  const __bf16* wr = Wo + tid * 512;
  for (int k = 0; k < 512; ++k) acc += v[k] * (float)wr[k];
  out[((size_t)b * WOUT + step) * 64 + tid] = acc;
  prev_y[b * 64 + tid] = acc;
}

// ---------------- prep kernels ----------------
__global__ void f2bf_kernel(const float* __restrict__ s, __bf16* __restrict__ d, int n) {
  for (int i = blockIdx.x * blockDim.x + threadIdx.x; i < n; i += gridDim.x * blockDim.x)
    d[i] = (__bf16)s[i];
}
__global__ void wcat_kernel(const float* __restrict__ A, int ka,
                            const float* __restrict__ B2, int kb,
                            __bf16* __restrict__ d, int rows) {
  const int kt = ka + kb, n = rows * kt;
  for (int i = blockIdx.x * blockDim.x + threadIdx.x; i < n; i += gridDim.x * blockDim.x) {
    const int r = i / kt, k = i - r * kt;
    const float v = (k < ka) ? A[(size_t)r * ka + k] : B2[(size_t)r * kb + (k - ka)];
    d[i] = (__bf16)v;
  }
}
__global__ void vadd_kernel(const float* a, const float* b, float* o, int n) {
  for (int i = blockIdx.x * blockDim.x + threadIdx.x; i < n; i += gridDim.x * blockDim.x)
    o[i] = a[i] + b[i];
}
__global__ void init_state_kernel(__bf16* h0, __bf16* h1, float* c0, float* c1,
                                  const float* __restrict__ x_in, float* prev_y) {
  const int N = 2 * Bsz * Hd;  // ping-pong pair
  for (int i = blockIdx.x * blockDim.x + threadIdx.x; i < N; i += gridDim.x * blockDim.x) {
    h0[i] = (__bf16)0.f;
    h1[i] = (__bf16)0.f;
    if (i < Bsz * Hd) { c0[i] = 0.f; c1[i] = 0.f; }
    if (i < Bsz * Din) {
      const int b = i >> 6, d = i & 63;
      prev_y[i] = x_in[((size_t)b * Tlen + (Tlen - 1)) * Din + d];
    }
  }
}

// =====================================================================
extern "C" void kernel_launch(void* const* d_in, const int* in_sizes, int n_in,
                              void* d_out, int out_size, void* d_ws, size_t ws_size,
                              hipStream_t stream) {
  const float* x_in  = (const float*)d_in[0];
  // d_in[1] = w_out (96), compile-time below
  const float* Wih0 = (const float*)d_in[2];
  const float* Whh0 = (const float*)d_in[3];
  const float* bih0 = (const float*)d_in[4];
  const float* bhh0 = (const float*)d_in[5];
  const float* Wih1 = (const float*)d_in[6];
  const float* Whh1 = (const float*)d_in[7];
  const float* bih1 = (const float*)d_in[8];
  const float* bhh1 = (const float*)d_in[9];
  const float* Wihd = (const float*)d_in[10];
  const float* Whhd = (const float*)d_in[11];
  const float* bihd = (const float*)d_in[12];
  const float* bhhd = (const float*)d_in[13];
  const float* Wp   = (const float*)d_in[14];
  const float* bp   = (const float*)d_in[15];
  const float* Wo   = (const float*)d_in[16];
  const float* bo   = (const float*)d_in[17];
  float* out = (float*)d_out;

  char* ws = (char*)d_ws;
  size_t off = 0;
  auto alloc = [&](size_t bytes) -> char* {
    char* p = ws + off;
    off += (bytes + 255) & ~(size_t)255;
    return p;
  };
  __bf16* x_bf    = (__bf16*)alloc((size_t)Bsz * Tlen * Din * 2);   // 16 MB
  __bf16* enc_out = (__bf16*)alloc((size_t)Bsz * Tlen * Hd * 2);    // 64 MB
  __bf16* W0c  = (__bf16*)alloc((size_t)G4 * (Din + Hd) * 2);
  __bf16* W1c  = (__bf16*)alloc((size_t)G4 * (2 * Hd) * 2);
  __bf16* Wdc  = (__bf16*)alloc((size_t)G4 * (3 * Hd) * 2);
  __bf16* Wpb  = (__bf16*)alloc((size_t)Hd * Din * 2);
  __bf16* Wob  = (__bf16*)alloc((size_t)Din * 2 * Hd * 2);
  float*  b0   = (float*)alloc(G4 * 4);
  float*  b1   = (float*)alloc(G4 * 4);
  float*  bd   = (float*)alloc(G4 * 4);
  __bf16* h0pp = (__bf16*)alloc((size_t)2 * Bsz * Hd * 2);   // ping-pong pair
  __bf16* h1pp = (__bf16*)alloc((size_t)2 * Bsz * Hd * 2);
  float*  c0   = (float*)alloc((size_t)Bsz * Hd * 4);
  float*  c1   = (float*)alloc((size_t)Bsz * Hd * 4);
  float*  prev_y = (float*)alloc((size_t)Bsz * Din * 4);
  __bf16* inp_bf = (__bf16*)alloc((size_t)Bsz * 2 * Hd * 2);
  (void)ws_size; (void)in_sizes; (void)n_in; (void)out_size;

  __bf16* h0b[2] = {h0pp, h0pp + (size_t)Bsz * Hd};
  __bf16* h1b[2] = {h1pp, h1pp + (size_t)Bsz * Hd};

  // ---- prep (runs every call; deterministic) ----
  f2bf_kernel<<<2048, 256, 0, stream>>>(x_in, x_bf, Bsz * Tlen * Din);
  wcat_kernel<<<512, 256, 0, stream>>>(Wih0, Din, Whh0, Hd, W0c, G4);
  wcat_kernel<<<512, 256, 0, stream>>>(Wih1, Hd, Whh1, Hd, W1c, G4);
  wcat_kernel<<<768, 256, 0, stream>>>(Wihd, 2 * Hd, Whhd, Hd, Wdc, G4);
  f2bf_kernel<<<64, 256, 0, stream>>>(Wp, Wpb, Hd * Din);
  f2bf_kernel<<<128, 256, 0, stream>>>(Wo, Wob, Din * 2 * Hd);
  vadd_kernel<<<4, 256, 0, stream>>>(bih0, bhh0, b0, G4);
  vadd_kernel<<<4, 256, 0, stream>>>(bih1, bhh1, b1, G4);
  vadd_kernel<<<4, 256, 0, stream>>>(bihd, bhhd, bd, G4);
  init_state_kernel<<<512, 256, 0, stream>>>(h0pp, h1pp, c0, c1, x_in, prev_y);

  // ---- encoder: 513 pipelined steps (layer0 at t=s, layer1 at t=s-1) ----
  dim3 gE(Hd / 32, Bsz / 64, 2);
  for (int s = 0; s <= Tlen; ++s) {
    encoder_step_kernel<<<gE, 128, 0, stream>>>(
        s, x_bf,
        h0b[(s + 1) & 1], h0b[s & 1],
        h1b[(s + 1) & 1], h1b[s & 1],
        c0, c1, enc_out, W0c, b0, W1c, b1);
  }
  // encoder final h1/c1 state: h1b[0] (s=512 writes parity 0), c1 in place.

  // ---- decoder: 96 steps x {attention, gates+cell, output} ----
  dim3 gG(Hd / 32, Bsz / 64, 1);
  for (int d = 0; d < WOUT; ++d) {
    attn_kernel<<<Bsz, 256, 0, stream>>>(enc_out, h1b[d & 1], prev_y, Wpb, bp, inp_bf);
    decoder_gates_kernel<<<gG, 128, 0, stream>>>(inp_bf, h1b[d & 1], h1b[(d + 1) & 1], c1, Wdc, bd);
    out_y_kernel<<<Bsz, 64, 0, stream>>>(h1b[(d + 1) & 1], inp_bf, Wob, bo, out, d, prev_y);
  }
}
